// TemporalWaveletInceptionModule_44315472560740
// MI455X (gfx1250) — matrix operand order repaired and checked by reading, not compiled, not run on hardware
//
#include <hip/hip_runtime.h>
#include <hip/hip_bf16.h>
#include <math.h>

// ---------------------------------------------------------------------------
// Types for gfx1250 WMMA
// ---------------------------------------------------------------------------
typedef __attribute__((ext_vector_type(16))) __bf16 v16bf;
typedef __attribute__((ext_vector_type(8)))  __bf16 v8bf;
typedef __attribute__((ext_vector_type(2)))  __bf16 v2bf;
typedef __attribute__((ext_vector_type(8)))  float  v8f;

// db8-style decomposition low-pass filter (band length 16)
__constant__ float c_dec_lo[16] = {
  -0.00011747678400228192f,  0.0006754494059985568f, -0.0003917403729959771f,
  -0.00487035299301066f,     0.008746094047015655f,   0.013981027917015516f,
  -0.04408825393106472f,    -0.01736930100202211f,    0.128747426620186f,
   0.00047248457399797254f, -0.2840155429624281f,    -0.015829105256023893f,
   0.5853546836548691f,      0.6756307362980128f,     0.3128715909144659f,
   0.05441584224308161f
};

// ---------------------------------------------------------------------------
// Banded DWT matrix generation (+ transposed copies)
// ---------------------------------------------------------------------------
__global__ void band_init_k(float* lo, float* hi, float* loT, float* hiT, int S) {
  int idx = blockIdx.x * blockDim.x + threadIdx.x;
  int half = S >> 1;
  if (idx >= half * S) return;
  int i = idx / S, j = idx - i * S;
  int t = j + 7 - 2 * i;
  float vlo = 0.f, vhi = 0.f;
  if (t >= 0 && t < 16) {
    vlo = c_dec_lo[15 - t];
    float d = c_dec_lo[t];
    vhi = (t & 1) ? -d : d;
  }
  lo[idx] = vlo;
  hi[idx] = vhi;
  loT[(size_t)j * half + i] = vlo;
  hiT[(size_t)j * half + i] = vhi;
}

// out_wT[k][n] = out_w[n][k]  (D x D)
__global__ void transpose_w_k(const float* __restrict__ w, float* __restrict__ wT, int D) {
  int idx = blockIdx.x * blockDim.x + threadIdx.x;
  if (idx >= D * D) return;
  int k = idx / D, n = idx - k * D;
  wT[idx] = w[(size_t)n * D + k];
}

// xd[(b*N+n)][t][d] = x[b][t][n][d]  (vectorized over d, D % 4 == 0)
__global__ void transpose_x_k(const float4* __restrict__ x, float4* __restrict__ xd,
                              int B, int T, int N, int D) {
  size_t idx = (size_t)blockIdx.x * blockDim.x + threadIdx.x;
  int D4 = D >> 2;
  size_t total = (size_t)B * N * T * D4;
  if (idx >= total) return;
  int d4 = idx % D4;
  size_t t1 = idx / D4;
  int t = t1 % T;
  size_t img = t1 / T;
  int n = img % N, b = img / N;
  xd[idx] = x[(((size_t)b * T + t) * N + n) * D4 + d4];
}

// out[b][t][n][d] = y[(b*N+n)][t][d]
__global__ void scatter_out_k(const float4* __restrict__ y, float4* __restrict__ out,
                              int B, int T, int N, int D) {
  size_t idx = (size_t)blockIdx.x * blockDim.x + threadIdx.x;
  int D4 = D >> 2;
  size_t total = (size_t)B * N * T * D4;
  if (idx >= total) return;
  int d4 = idx % D4;
  size_t t1 = idx / D4;
  int t = t1 % T;
  size_t img = t1 / T;
  int n = img % N, b = img / N;
  out[(((size_t)b * T + t) * N + n) * D4 + d4] = y[idx];
}

// ---------------------------------------------------------------------------
// Batched GEMM with v_wmma_f32_16x16x32_bf16.
//   C[b] = A[b] @ B[b] (+ C[b] if beta) (+ bias[col] if bias)
// fp32 operands converted to bf16 during LDS staging; LDS holds packed bf16:
//   As : row-major  [m][k]  pitch APITCH, fragment = two 16B chunks
//   Bt : transposed [n][k]  pitch BPITCH, fragment = one 32B run
// Block = 256 threads (8 wave32), 64x64 macro tile, K-step 32.
// Uniform fast path for fully in-bounds tiles (no per-lane guards).
// ---------------------------------------------------------------------------
#define TM 64
#define TN 64
#define TK 32
#define APITCH 40   // bf16 elems per A row (64B data + 16B pad) -> conflict-free
#define BPITCH 40

__global__ __launch_bounds__(256)
void gemm_wmma_k(const float* __restrict__ A, long long sA, int lda,
                 const float* __restrict__ B, long long sB, int ldb,
                 float* __restrict__ C, long long sC, int ldc,
                 int M, int N, int K, int beta, const float* __restrict__ bias)
{
  __shared__ __align__(16) __bf16 As[TM * APITCH];   // 5120 B
  __shared__ __align__(16) __bf16 Bt[TN * BPITCH];   // 5120 B

  const int bz = blockIdx.z;
  A += (long long)bz * sA;
  B += (long long)bz * sB;
  C += (long long)bz * sC;

  const int m0 = blockIdx.x * TM;
  const int n0 = blockIdx.y * TN;
  const int tid  = threadIdx.x;
  const int wave = tid >> 5;
  const int lane = tid & 31;
  const int laneM   = lane & 15;
  const int laneHlf = lane >> 4;

  // wave -> output subtiles: one M tile, two adjacent N tiles (shared A frag)
  const int mt  = wave >> 1;          // 0..3
  const int nt0 = (wave & 1) * 2;     // 0 or 2

  v8f acc0 = {};
  v8f acc1 = {};

  const bool mnFull = (m0 + TM <= M) && (n0 + TN <= N);

  for (int k0 = 0; k0 < K; k0 += TK) {
    const bool full = mnFull && (k0 + TK <= K);
    if (full) {
      // ---- fast path: unguarded float2 loads, packed bf16 stores ----
      #pragma unroll
      for (int it = 0; it < 4; ++it) {          // A: 64 x 16 pairs
        int idx = it * 256 + tid;
        int k2 = idx & 15, m = idx >> 4;
        float2 va = *(const float2*)(A + (size_t)(m0 + m) * lda + k0 + 2 * k2);
        v2bf p; p[0] = (__bf16)va.x; p[1] = (__bf16)va.y;
        *(v2bf*)&As[m * APITCH + 2 * k2] = p;
      }
      #pragma unroll
      for (int it = 0; it < 4; ++it) {          // B: 64 cols x 16 pairs, transposed
        int idx = it * 256 + tid;
        int n = idx & 63, k2 = idx >> 6;
        const float* bp = B + (size_t)(k0 + 2 * k2) * ldb + n0 + n;
        float x0 = bp[0];
        float x1 = bp[ldb];
        v2bf p; p[0] = (__bf16)x0; p[1] = (__bf16)x1;
        *(v2bf*)&Bt[n * BPITCH + 2 * k2] = p;
      }
    } else {
      // ---- guarded path for edge tiles ----
      #pragma unroll
      for (int it = 0; it < 4; ++it) {
        int idx = it * 256 + tid;
        int k2 = idx & 15, m = idx >> 4;
        int gm = m0 + m, gk = k0 + 2 * k2;
        float x0 = 0.f, x1 = 0.f;
        if (gm < M) {
          if (gk < K)     x0 = A[(size_t)gm * lda + gk];
          if (gk + 1 < K) x1 = A[(size_t)gm * lda + gk + 1];
        }
        v2bf p; p[0] = (__bf16)x0; p[1] = (__bf16)x1;
        *(v2bf*)&As[m * APITCH + 2 * k2] = p;
      }
      #pragma unroll
      for (int it = 0; it < 4; ++it) {
        int idx = it * 256 + tid;
        int n = idx & 63, k2 = idx >> 6;
        int gn = n0 + n, gk = k0 + 2 * k2;
        float x0 = 0.f, x1 = 0.f;
        if (gn < N) {
          if (gk < K)     x0 = B[(size_t)gk * ldb + gn];
          if (gk + 1 < K) x1 = B[(size_t)(gk + 1) * ldb + gn];
        }
        v2bf p; p[0] = (__bf16)x0; p[1] = (__bf16)x1;
        *(v2bf*)&Bt[n * BPITCH + 2 * k2] = p;
      }
    }
    if (k0 + TK < K) {   // global_prefetch_b8 of next K tile
      __builtin_prefetch(A + (size_t)(m0 + (tid >> 2)) * lda + k0 + TK, 0, 1);
      __builtin_prefetch(B + (size_t)(k0 + TK + (tid >> 4)) * ldb + n0, 0, 1);
    }
    __syncthreads();

    // ---- fragments: contiguous packed-bf16 LDS reads ----
    // A: elems e -> K = half*16 + e at fixed row
    const __bf16* ap = &As[(mt * 16 + laneM) * APITCH + laneHlf * 16];
    v8bf aLo = *(const v8bf*)ap;
    v8bf aHi = *(const v8bf*)(ap + 8);
    // B: elems e -> K = half*16 + e at fixed column
    const __bf16* bp0 = &Bt[(nt0 * 16 + laneM) * BPITCH + laneHlf * 16];
    v8bf b0Lo = *(const v8bf*)bp0;
    v8bf b0Hi = *(const v8bf*)(bp0 + 8);
    const __bf16* bp1 = bp0 + 16 * BPITCH;
    v8bf b1Lo = *(const v8bf*)bp1;
    v8bf b1Hi = *(const v8bf*)(bp1 + 8);

    v16bf fa, fb0, fb1;
    #pragma unroll
    for (int e = 0; e < 8; ++e) {
      fa[e] = aLo[e];   fa[e + 8] = aHi[e];
      fb0[e] = b0Lo[e]; fb0[e + 8] = b0Hi[e];
      fb1[e] = b1Lo[e]; fb1[e + 8] = b1Hi[e];
    }

    acc0 = __builtin_amdgcn_wmma_f32_16x16x32_bf16(
        false, fa, false, fb0, (short)0, acc0, false, false);
    acc1 = __builtin_amdgcn_wmma_f32_16x16x32_bf16(
        false, fa, false, fb1, (short)0, acc1, false, false);
    __syncthreads();
  }

  // ---- epilogue: C/D layout VGPR r -> M = r + (lane/16)*8, N = lane%16 ----
  #pragma unroll
  for (int r = 0; r < 8; ++r) {
    int row  = m0 + mt * 16 + r + laneHlf * 8;
    int col0 = n0 + nt0 * 16 + laneM;
    int col1 = col0 + 16;
    if (row < M) {
      if (col0 < N) {
        float v = acc0[r];
        if (beta) v += C[(size_t)row * ldc + col0];
        if (bias) v += bias[col0];
        C[(size_t)row * ldc + col0] = v;
      }
      if (col1 < N) {
        float v = acc1[r];
        if (beta) v += C[(size_t)row * ldc + col1];
        if (bias) v += bias[col1];
        C[(size_t)row * ldc + col1] = v;
      }
    }
  }
}

// ---------------------------------------------------------------------------
// Per-row median-low of |HH| -> threshold; soft-threshold LH/HL/HH.
// ---------------------------------------------------------------------------
__global__ void median_soft_k(const float* __restrict__ HH,
                              const float* __restrict__ LH,
                              const float* __restrict__ HL,
                              float* __restrict__ sHH,
                              float* __restrict__ sLH,
                              float* __restrict__ sHL,
                              int rows, int w, float s2logT)
{
  extern __shared__ float rowbuf[];
  int r = blockIdx.x * blockDim.x + threadIdx.x;
  float* my = rowbuf + (size_t)threadIdx.x * w;
  if (r >= rows) return;
  const size_t base = (size_t)r * w;
  for (int j = 0; j < w; ++j) my[j] = fabsf(HH[base + j]);
  const int k = (w - 1) >> 1;
  float med = 0.f;
  for (int j = 0; j < w; ++j) {
    float v = my[j];
    int less = 0, leq = 0;
    for (int t = 0; t < w; ++t) {
      less += (my[t] < v);
      leq  += (my[t] <= v);
    }
    if (less <= k && k < leq) { med = v; break; }
  }
  const float thr = med / 0.6745f * s2logT;
  for (int j = 0; j < w; ++j) {
    float v, a, s;
    v = LH[base + j]; a = fabsf(v) - thr; s = (v > 0.f) - (v < 0.f);
    sLH[base + j] = (a > 0.f ? a : 0.f) * s;
    v = HL[base + j]; a = fabsf(v) - thr; s = (v > 0.f) - (v < 0.f);
    sHL[base + j] = (a > 0.f ? a : 0.f) * s;
    v = HH[base + j]; a = fabsf(v) - thr; s = (v > 0.f) - (v < 0.f);
    sHH[base + j] = (a > 0.f ? a : 0.f) * s;
  }
}

// ---------------------------------------------------------------------------
// Inception: mean over 6 SAME convs (k = 1,3,5,7,9,11), C=1.
// 16x16 output tile per block; 26x26 halo tile staged to LDS with the
// gfx1250 async copy path (global_load_async_to_lds_b32 + s_wait_asynccnt).
// LDS pre-zeroed so the halo implements SAME zero padding.
// ---------------------------------------------------------------------------
#define ITILE 26          // 16 + 2*5 halo
#define IPITCH 27

__global__ __launch_bounds__(256)
void inception_k(const float* __restrict__ in, float* __restrict__ out,
                 const float* __restrict__ w0, const float* __restrict__ w1,
                 const float* __restrict__ w2, const float* __restrict__ w3,
                 const float* __restrict__ w4, const float* __restrict__ w5,
                 const float* __restrict__ b0, const float* __restrict__ b1,
                 const float* __restrict__ b2, const float* __restrict__ b3,
                 const float* __restrict__ b4, const float* __restrict__ b5,
                 int h, int w)
{
  __shared__ float wsm[286];
  __shared__ __align__(16) float tile[ITILE * IPITCH];

  const float* wp[6] = {w0, w1, w2, w3, w4, w5};
  const int off[7] = {0, 1, 10, 35, 84, 165, 286};
  for (int i = threadIdx.x; i < 286; i += 256) {
    int ki = 0;
    while (i >= off[ki + 1]) ki++;
    wsm[i] = wp[ki][i - off[ki]];
  }
  // zero the halo tile
  for (int i = threadIdx.x; i < ITILE * IPITCH; i += 256) tile[i] = 0.f;
  __syncthreads();

  const int x0 = blockIdx.x * 16;
  const int y0 = blockIdx.y * 16;
  const float* im = in + (size_t)blockIdx.z * h * w;

  // async-stage 26x26 input window (in-range lanes only; EXEC masks the rest)
  for (int i = threadIdx.x; i < ITILE * ITILE; i += 256) {
    int ty = i / ITILE, tx = i - ty * ITILE;
    int gy = y0 - 5 + ty, gx = x0 - 5 + tx;
    if (gy >= 0 && gy < h && gx >= 0 && gx < w) {
      unsigned lds_off = (unsigned)(size_t)&tile[ty * IPITCH + tx];
      const float* gp = im + (size_t)gy * w + gx;
      asm volatile("global_load_async_to_lds_b32 %0, %1, off"
                   :: "v"(lds_off), "v"(gp) : "memory");
    }
  }
  asm volatile("s_wait_asynccnt 0x0" ::: "memory");
  __syncthreads();

  const float bsum = *b0 + *b1 + *b2 + *b3 + *b4 + *b5;
  const int lx = threadIdx.x & 15;
  const int ly = threadIdx.x >> 4;
  const int ox = x0 + lx, oy = y0 + ly;
  if (ox >= w || oy >= h) return;

  float s = 0.f;
  #pragma unroll
  for (int ki = 0; ki < 6; ++ki) {
    const int ksz = 2 * ki + 1;
    const int o = 5 - ki;
    const float* wkp = wsm + off[ki];
    float a = 0.f;
    for (int dy = 0; dy < ksz; ++dy)
      for (int dx = 0; dx < ksz; ++dx)
        a += wkp[dy * ksz + dx] * tile[(ly + o + dy) * IPITCH + (lx + o + dx)];
    s += a;
  }
  out[((size_t)blockIdx.z * h + oy) * w + ox] = (s + bsum) * (1.0f / 6.0f);
}

// ---------------------------------------------------------------------------
// Host-side orchestration
// ---------------------------------------------------------------------------
extern "C" void kernel_launch(void* const* d_in, const int* in_sizes, int n_in,
                              void* d_out, int out_size, void* d_ws, size_t ws_size,
                              hipStream_t stream)
{
  (void)in_sizes; (void)n_in; (void)out_size; (void)ws_size;
  const int B = 8, T = 128, N = 100, D = 256;
  const int IMG = B * N;            // 800
  const float s2logT = sqrtf(2.0f * logf((float)T));

  // setup_inputs order: x, adj, (w0,b0) ... (w5,b5), out_w, out_b
  const float* x     = (const float*)d_in[0];
  const float* wk[6] = {(const float*)d_in[2], (const float*)d_in[4],
                        (const float*)d_in[6], (const float*)d_in[8],
                        (const float*)d_in[10], (const float*)d_in[12]};
  const float* bk[6] = {(const float*)d_in[3], (const float*)d_in[5],
                        (const float*)d_in[7], (const float*)d_in[9],
                        (const float*)d_in[11], (const float*)d_in[13]};
  const float* out_w = (const float*)d_in[14];
  const float* out_b = (const float*)d_in[15];
  float* out = (float*)d_out;

  // ---- workspace layout (floats) ----
  float* wsf = (float*)d_ws;
  size_t off = 0;
  auto alloc = [&](size_t n) { float* p = wsf + off; off += n; return p; };

  const int sizes[4] = {32, 64, 128, 256};
  float *loM[4], *hiM[4], *loTM[4], *hiTM[4];
  for (int s = 0; s < 4; ++s) {
    size_t hs = (size_t)(sizes[s] / 2) * sizes[s];
    loM[s] = alloc(hs);  hiM[s] = alloc(hs);
    loTM[s] = alloc(hs); hiTM[s] = alloc(hs);
  }
  auto bi = [&](int S) { return S == 32 ? 0 : S == 64 ? 1 : S == 128 ? 2 : 3; };

  float* owT   = alloc((size_t)D * D);
  float* xd    = alloc((size_t)IMG * T * D);        // also idwt final output
  float* Lp    = alloc((size_t)IMG * 64 * 256);     // row-pass temp / idwt L
  float* Hp    = alloc((size_t)IMG * 64 * 256);     // row-pass temp / idwt H
  float* LLraw = alloc((size_t)IMG * 64 * 128);     // start of 4-temp block (=ybuf)
  float* sLH   = alloc((size_t)IMG * 64 * 128);
  float* sHL   = alloc((size_t)IMG * 64 * 128);
  float* sHH   = alloc((size_t)IMG * 64 * 128);
  float *LHd[3], *HLd[3], *HHd[3];
  for (int i = 0; i < 3; ++i) {
    size_t sz = (size_t)IMG * (64 >> i) * (128 >> i);
    LHd[i] = alloc(sz); HLd[i] = alloc(sz); HHd[i] = alloc(sz);
  }
  float* ll0 = alloc((size_t)IMG * 64 * 128);
  float* ll1 = alloc((size_t)IMG * 32 * 64);
  float* ll2 = alloc((size_t)IMG * 16 * 32);
  float* ybuf = LLraw;

  // ---- launchers ----
  auto gemm = [&](const float* A, long long sA, int lda,
                  const float* Bp, long long sB, int ldb,
                  float* C, long long sC, int ldc,
                  int M, int Nn, int K, int beta, const float* bias, int batch) {
    dim3 g((M + TM - 1) / TM, (Nn + TN - 1) / TN, batch);
    gemm_wmma_k<<<g, dim3(256), 0, stream>>>(A, sA, lda, Bp, sB, ldb,
                                             C, sC, ldc, M, Nn, K, beta, bias);
  };
  auto incep = [&](const float* in, float* o, int batch, int h, int w) {
    dim3 g((w + 15) / 16, (h + 15) / 16, batch);
    inception_k<<<g, dim3(256), 0, stream>>>(
        in, o, wk[0], wk[1], wk[2], wk[3], wk[4], wk[5],
        bk[0], bk[1], bk[2], bk[3], bk[4], bk[5], h, w);
  };

  // ---- init: band matrices, weight transpose, input transpose ----
  for (int s = 0; s < 4; ++s) {
    int S = sizes[s];
    int n = (S / 2) * S;
    band_init_k<<<dim3((n + 255) / 256), dim3(256), 0, stream>>>(
        loM[s], hiM[s], loTM[s], hiTM[s], S);
  }
  transpose_w_k<<<dim3((D * D + 255) / 256), dim3(256), 0, stream>>>(out_w, owT, D);
  {
    size_t total = (size_t)IMG * T * (D / 4);
    transpose_x_k<<<dim3((total + 255) / 256), dim3(256), 0, stream>>>(
        (const float4*)x, (float4*)xd, B, T, N, D);
  }

  // ---- forward DWT levels ----
  const float* LLin[3] = {xd, ll0, ll1};
  float* llOut[3] = {ll0, ll1, ll2};
  for (int i = 0; i < 3; ++i) {
    int Hn = 128 >> i, Wn = 256 >> i, h = Hn >> 1, w = Wn >> 1;
    long long sIn = (long long)Hn * Wn, sRow = (long long)h * Wn,
              sSub = (long long)h * w;
    gemm(loM[bi(Hn)], 0, Hn, LLin[i], sIn, Wn, Lp, sRow, Wn, h, Wn, Hn, 0, nullptr, IMG);
    gemm(hiM[bi(Hn)], 0, Hn, LLin[i], sIn, Wn, Hp, sRow, Wn, h, Wn, Hn, 0, nullptr, IMG);
    gemm(Lp, sRow, Wn, loTM[bi(Wn)], 0, w, LLraw,  sSub, w, h, w, Wn, 0, nullptr, IMG);
    gemm(Lp, sRow, Wn, hiTM[bi(Wn)], 0, w, LHd[i], sSub, w, h, w, Wn, 0, nullptr, IMG);
    gemm(Hp, sRow, Wn, loTM[bi(Wn)], 0, w, HLd[i], sSub, w, h, w, Wn, 0, nullptr, IMG);
    gemm(Hp, sRow, Wn, hiTM[bi(Wn)], 0, w, HHd[i], sSub, w, h, w, Wn, 0, nullptr, IMG);
    int rows = IMG * h;
    median_soft_k<<<dim3((rows + 63) / 64), dim3(64), 64 * w * sizeof(float), stream>>>(
        HHd[i], LHd[i], HLd[i], sHH, sLH, sHL, rows, w, s2logT);
    incep(LLraw, llOut[i], IMG, h, w);
    incep(sLH, LHd[i], IMG, h, w);
    incep(sHL, HLd[i], IMG, h, w);
    incep(sHH, HHd[i], IMG, h, w);
  }

  // ---- inverse DWT levels ----
  const float* LLcur[3] = {ll2, ll1, ll0};     // j = 2, 1, 0
  float* rec[3] = {ll1, ll0, xd};
  for (int jj = 0; jj < 3; ++jj) {
    int j = 2 - jj;
    int Hn = 128 >> j, Wn = 256 >> j, h = Hn >> 1, w = Wn >> 1;
    long long sSub = (long long)h * w, sRow = (long long)h * Wn,
              sOut = (long long)Hn * Wn;
    gemm(LLcur[jj], sSub, w, loM[bi(Wn)], 0, Wn, Lp, sRow, Wn, h, Wn, w, 0, nullptr, IMG);
    gemm(LHd[j],    sSub, w, hiM[bi(Wn)], 0, Wn, Lp, sRow, Wn, h, Wn, w, 1, nullptr, IMG);
    gemm(HLd[j],    sSub, w, loM[bi(Wn)], 0, Wn, Hp, sRow, Wn, h, Wn, w, 0, nullptr, IMG);
    gemm(HHd[j],    sSub, w, hiM[bi(Wn)], 0, Wn, Hp, sRow, Wn, h, Wn, w, 1, nullptr, IMG);
    gemm(loTM[bi(Hn)], 0, h, Lp, sRow, Wn, rec[jj], sOut, Wn, Hn, Wn, h, 0, nullptr, IMG);
    gemm(hiTM[bi(Hn)], 0, h, Hp, sRow, Wn, rec[jj], sOut, Wn, Hn, Wn, h, 1, nullptr, IMG);
  }

  // ---- final projection y = LL @ out_w^T + out_b, then scatter ----
  gemm(xd, (long long)T * D, D, owT, 0, D, ybuf, (long long)T * D, D,
       T, D, D, 0, out_b, IMG);
  {
    size_t total = (size_t)IMG * T * (D / 4);
    scatter_out_k<<<dim3((total + 255) / 256), dim3(256), 0, stream>>>(
        (const float4*)ybuf, (float4*)out, B, T, N, D);
  }
}
